// FlashACE_46205258170441
// MI455X (gfx1250) — matrix operand order repaired and checked by reading, not compiled, not run on hardware
//
#include <hip/hip_runtime.h>
#include <hip/hip_bf16.h>

#define DHID 128

typedef __attribute__((ext_vector_type(16))) _Float16 v16h;
typedef __attribute__((ext_vector_type(8)))  float    v8f;

#define V8F_ZERO {0.f,0.f,0.f,0.f,0.f,0.f,0.f,0.f}
#define WMMA_F16(acc,a,b) (acc) = __builtin_amdgcn_wmma_f32_16x16x32_f16(false,(a),false,(b),(short)0,(acc),false,false)

__device__ __forceinline__ float silu_f(float x){ return x * (1.0f/(1.0f + __expf(-x))); }

// A-fragment (16x32 f16) per ISA layout: lanes 0-15 hold K={0..7,16..23},
// lanes 16-31 hold K={8..15,24..31}. rowp = LDS row ptr at the chunk's column.
__device__ __forceinline__ v16h ldA(const _Float16* rowp, int hi){
  v16h a; const int b = hi ? 8 : 0;
  #pragma unroll
  for (int j = 0; j < 8; ++j){ a[j] = rowp[b + j]; a[8 + j] = rowp[16 + b + j]; }
  return a;
}

// B-fragment: pre-swizzled in workspace, each lane's 16 halves contiguous (32B).
__device__ __forceinline__ v16h ldB(const _Float16* wf, int frag, int lane){
  const _Float16* p = wf + (((size_t)frag * 32 + (size_t)lane) << 4);
  v16h b;
  #pragma unroll
  for (int j = 0; j < 16; ++j) b[j] = p[j];
  return b;
}

// copy 8 consecutive f32 -> 8 f16
__device__ __forceinline__ void row8_f16(_Float16* dst, const float* src){
  float4 v0 = ((const float4*)src)[0];
  float4 v1 = ((const float4*)src)[1];
  dst[0]=(_Float16)v0.x; dst[1]=(_Float16)v0.y; dst[2]=(_Float16)v0.z; dst[3]=(_Float16)v0.w;
  dst[4]=(_Float16)v1.x; dst[5]=(_Float16)v1.y; dst[6]=(_Float16)v1.z; dst[7]=(_Float16)v1.w;
}

// ---------------------------------------------------------------------------
// Weight conversion: f32 (K x 128, row-major) -> f16 WMMA-B fragments.
// Fragment linear layout: idx = ((kc*8 + ct)*32 + lane)*16 + j
//   k = kc*32 + (j<8 ? 0 : 16) + (lane>=16 ? 8 : 0) + (j&7) ; n = ct*16 + (lane&15) ; pad rows -> 0
// ---------------------------------------------------------------------------
__global__ void wconv_kernel(const float* __restrict__ src, _Float16* __restrict__ dst,
                             int Kact, int total){
  int idx = blockIdx.x * blockDim.x + threadIdx.x;
  if (idx >= total) return;
  int j    = idx & 15;
  int lane = (idx >> 4) & 31;
  int frag = idx >> 9;
  int ct   = frag & 7;
  int kc   = frag >> 3;
  int n = ct * 16 + (lane & 15);
  int k = kc * 32 + ((j < 8) ? 0 : 16) + ((lane >= 16) ? 8 : 0) + (j & 7);
  float v = (k < Kact) ? src[(size_t)k * DHID + n] : 0.0f;
  dst[idx] = (_Float16)v;
}

// ---------------------------------------------------------------------------
// Edge-init: edge_state = mlp2([hs, hr, el])   (K = 257 -> 9 chunks of 32)
// block = 256 thr = 8 waves; 16 edges per block; wave w -> cols [16w,16w+16)
// ---------------------------------------------------------------------------
__global__ __launch_bounds__(256) void edge_init_kernel(
    const float* __restrict__ h_cur, float* __restrict__ es,
    const int* __restrict__ snd, const int* __restrict__ rcv,
    const float* __restrict__ elen,
    const _Float16* __restrict__ w1f, const float* __restrict__ b1,
    const _Float16* __restrict__ w2f, const float* __restrict__ b2,
    int E)
{
  __shared__ _Float16 Acm[16 * 288];
  __shared__ _Float16 HID[16 * 128];
  const int tid = threadIdx.x;
  const int lane = tid & 31, w = tid >> 5, hi = lane >> 4;
  const int e0 = blockIdx.x * 16;
  {
    int row = tid >> 4, seg = tid & 15;
    int e = e0 + row; if (e >= E) e = E - 1;
    int s = snd[e], r = rcv[e];
    row8_f16(&Acm[row*288 +       seg*8], h_cur + (size_t)s * DHID + seg*8);
    row8_f16(&Acm[row*288 + 128 + seg*8], h_cur + (size_t)r * DHID + seg*8);
    Acm[row*288 + 256 + 2*seg] = (seg == 0) ? (_Float16)elen[e] : (_Float16)0.0f;
    Acm[row*288 + 257 + 2*seg] = (_Float16)0.0f;
  }
  __syncthreads();

  const int m = lane & 15, col = w*16 + m, mb = hi * 8;

  v8f acc = V8F_ZERO;
  #pragma unroll
  for (int c = 0; c < 9; ++c){
    v16h a = ldA(&Acm[m*288 + c*32], hi);
    v16h b = ldB(w1f, c*8 + w, lane);
    WMMA_F16(acc, a, b);
  }
  { float bb = b1[col];
    #pragma unroll
    for (int r = 0; r < 8; ++r) HID[(mb + r)*128 + col] = (_Float16)silu_f(acc[r] + bb);
  }
  __syncthreads();
  v8f acc2 = V8F_ZERO;
  #pragma unroll
  for (int c = 0; c < 4; ++c){
    v16h a = ldA(&HID[m*128 + c*32], hi);
    v16h b = ldB(w2f, c*8 + w, lane);
    WMMA_F16(acc2, a, b);
  }
  { float bb = b2[col];
    #pragma unroll
    for (int r = 0; r < 8; ++r){
      int e = e0 + mb + r;
      if (e < E) es[(size_t)e * DHID + col] = acc2[r] + bb;
    }
  }
}

// ---------------------------------------------------------------------------
// Per-layer fused edge kernel:
//   es' = mlp2([hs, hr, es, el])   (K = 385 -> 13 chunks)
//   msg = mlp2([hs, hr, el])       (K = 257 -> 9 chunks)
//   hacc[receiver] += msg          (global f32 atomics)
// ---------------------------------------------------------------------------
__global__ __launch_bounds__(256) void edge_layer_kernel(
    const float* __restrict__ h_cur, float* __restrict__ es, float* __restrict__ hacc,
    const int* __restrict__ snd, const int* __restrict__ rcv,
    const float* __restrict__ elen,
    const _Float16* __restrict__ euw1, const float* __restrict__ eub1,
    const _Float16* __restrict__ euw2, const float* __restrict__ eub2,
    const _Float16* __restrict__ mpw1, const float* __restrict__ mpb1,
    const _Float16* __restrict__ mpw2, const float* __restrict__ mpb2,
    int E)
{
  __shared__ _Float16 Acm[16 * 288];   // [hs | hr | el,0...]
  __shared__ _Float16 ESb[16 * 128];
  __shared__ _Float16 HID[16 * 128];
  const int tid = threadIdx.x;
  const int lane = tid & 31, w = tid >> 5, hi = lane >> 4;
  const int e0 = blockIdx.x * 16;
  {
    int row = tid >> 4, seg = tid & 15;
    int e = e0 + row; if (e >= E) e = E - 1;
    int s = snd[e], r = rcv[e];
    row8_f16(&Acm[row*288 +       seg*8], h_cur + (size_t)s * DHID + seg*8);
    row8_f16(&Acm[row*288 + 128 + seg*8], h_cur + (size_t)r * DHID + seg*8);
    row8_f16(&ESb[row*128 +       seg*8], es    + (size_t)e * DHID + seg*8);
    Acm[row*288 + 256 + 2*seg] = (seg == 0) ? (_Float16)elen[e] : (_Float16)0.0f;
    Acm[row*288 + 257 + 2*seg] = (_Float16)0.0f;
  }
  __syncthreads();

  const int m = lane & 15, col = w*16 + m, mb = hi * 8;

  // ---- edge-update MLP (K rows: 0-255 hs|hr, 256-383 es, 384 el, pad->416) ----
  v8f acc = V8F_ZERO;
  #pragma unroll
  for (int c = 0; c < 13; ++c){
    const _Float16* ap;
    if (c < 8)       ap = &Acm[m*288 + c*32];
    else if (c < 12) ap = &ESb[m*128 + (c - 8)*32];
    else             ap = &Acm[m*288 + 256];          // [el, zeros] matches zero-padded rows
    v16h a = ldA(ap, hi);
    v16h b = ldB(euw1, c*8 + w, lane);
    WMMA_F16(acc, a, b);
  }
  { float bb = eub1[col];
    #pragma unroll
    for (int r = 0; r < 8; ++r) HID[(mb + r)*128 + col] = (_Float16)silu_f(acc[r] + bb);
  }
  __syncthreads();
  v8f acc2 = V8F_ZERO;
  #pragma unroll
  for (int c = 0; c < 4; ++c){
    v16h a = ldA(&HID[m*128 + c*32], hi);
    v16h b = ldB(euw2, c*8 + w, lane);
    WMMA_F16(acc2, a, b);
  }
  { float bb = eub2[col];
    #pragma unroll
    for (int r = 0; r < 8; ++r){
      int e = e0 + mb + r;
      if (e < E) es[(size_t)e * DHID + col] = acc2[r] + bb;
    }
  }
  __syncthreads();   // HID reads complete before message MLP reuses it

  // ---- message MLP ----
  v8f acc3 = V8F_ZERO;
  #pragma unroll
  for (int c = 0; c < 9; ++c){
    v16h a = ldA(&Acm[m*288 + c*32], hi);
    v16h b = ldB(mpw1, c*8 + w, lane);
    WMMA_F16(acc3, a, b);
  }
  { float bb = mpb1[col];
    #pragma unroll
    for (int r = 0; r < 8; ++r) HID[(mb + r)*128 + col] = (_Float16)silu_f(acc3[r] + bb);
  }
  __syncthreads();
  v8f acc4 = V8F_ZERO;
  #pragma unroll
  for (int c = 0; c < 4; ++c){
    v16h a = ldA(&HID[m*128 + c*32], hi);
    v16h b = ldB(mpw2, c*8 + w, lane);
    WMMA_F16(acc4, a, b);
  }
  { float bb = mpb2[col];
    #pragma unroll
    for (int r = 0; r < 8; ++r){
      int e = e0 + mb + r;
      if (e < E){
        int rc = rcv[e];
        atomicAdd(&hacc[(size_t)rc * DHID + col], acc4[r] + bb);
      }
    }
  }
}

// ---------------------------------------------------------------------------
// Node update: h_io holds t = h + segment_sum(msgs); in place: t += mlp2(t)
// ---------------------------------------------------------------------------
__global__ __launch_bounds__(256) void node_kernel(
    float* __restrict__ h_io,
    const _Float16* __restrict__ w1f, const float* __restrict__ b1,
    const _Float16* __restrict__ w2f, const float* __restrict__ b2,
    int N)
{
  __shared__ _Float16 Tb[16 * 128];
  __shared__ _Float16 HID[16 * 128];
  const int tid = threadIdx.x;
  const int lane = tid & 31, w = tid >> 5, hi = lane >> 4;
  const int n0 = blockIdx.x * 16;
  {
    int row = tid >> 4, seg = tid & 15;
    int n = n0 + row; if (n >= N) n = N - 1;
    row8_f16(&Tb[row*128 + seg*8], h_io + (size_t)n * DHID + seg*8);
  }
  __syncthreads();

  const int m = lane & 15, col = w*16 + m, mb = hi * 8;

  v8f acc = V8F_ZERO;
  #pragma unroll
  for (int c = 0; c < 4; ++c){
    v16h a = ldA(&Tb[m*128 + c*32], hi);
    v16h b = ldB(w1f, c*8 + w, lane);
    WMMA_F16(acc, a, b);
  }
  { float bb = b1[col];
    #pragma unroll
    for (int r = 0; r < 8; ++r) HID[(mb + r)*128 + col] = (_Float16)silu_f(acc[r] + bb);
  }
  __syncthreads();
  v8f acc2 = V8F_ZERO;
  #pragma unroll
  for (int c = 0; c < 4; ++c){
    v16h a = ldA(&HID[m*128 + c*32], hi);
    v16h b = ldB(w2f, c*8 + w, lane);
    WMMA_F16(acc2, a, b);
  }
  { float bb = b2[col];
    #pragma unroll
    for (int r = 0; r < 8; ++r){
      int n = n0 + mb + r;
      if (n < N){
        size_t idx = (size_t)n * DHID + col;
        h_io[idx] = h_io[idx] + acc2[r] + bb;   // residual (block owns its rows)
      }
    }
  }
}

// ---------------------------------------------------------------------------
extern "C" void kernel_launch(void* const* d_in, const int* in_sizes, int n_in,
                              void* d_out, int out_size, void* d_ws, size_t ws_size,
                              hipStream_t stream)
{
  const float* h     = (const float*)d_in[0];
  const int*   eix   = (const int*)  d_in[1];
  const float* elen  = (const float*)d_in[2];
  const float* ei_w1 = (const float*)d_in[3];  const float* ei_b1 = (const float*)d_in[4];
  const float* ei_w2 = (const float*)d_in[5];  const float* ei_b2 = (const float*)d_in[6];
  const float* eu_w1 = (const float*)d_in[7];  const float* eu_b1 = (const float*)d_in[8];
  const float* eu_w2 = (const float*)d_in[9];  const float* eu_b2 = (const float*)d_in[10];
  const float* mp_w1 = (const float*)d_in[11]; const float* mp_b1 = (const float*)d_in[12];
  const float* mp_w2 = (const float*)d_in[13]; const float* mp_b2 = (const float*)d_in[14];
  const float* nu_w1 = (const float*)d_in[15]; const float* nu_b1 = (const float*)d_in[16];
  const float* nu_w2 = (const float*)d_in[17]; const float* nu_b2 = (const float*)d_in[18];

  const int N = in_sizes[0] / DHID;
  const int E = in_sizes[1] / 2;
  int L = in_sizes[7] / ((3 * DHID + 1) * DHID);   // eu_w1 is (L, 385, 128)
  if (L < 1) L = 1; if (L > 8) L = 8;

  const int* snd = eix;
  const int* rcv = eix + E;

  // ---- workspace layout: converted f16 weight fragments, then h ping buffer ----
  _Float16* wb = (_Float16*)d_ws;
  size_t off = 0;
  auto walloc = [&](int chunks)->size_t { size_t o = off; off += (size_t)chunks * 8 * 512; return o; };
  size_t o_eiw1 = walloc(9), o_eiw2 = walloc(4);
  size_t o_euw1[8], o_euw2[8], o_mpw1[8], o_mpw2[8], o_nuw1[8], o_nuw2[8];
  for (int l = 0; l < L; ++l){
    o_euw1[l] = walloc(13); o_euw2[l] = walloc(4);
    o_mpw1[l] = walloc(9);  o_mpw2[l] = walloc(4);
    o_nuw1[l] = walloc(4);  o_nuw2[l] = walloc(4);
  }
  size_t wbytes = (off * sizeof(_Float16) + 255) & ~(size_t)255;
  float* H1 = (float*)((char*)d_ws + wbytes);

  auto conv = [&](const float* src, size_t dsto, int Kact, int chunks){
    int total = chunks * 8 * 512;
    wconv_kernel<<<(total + 255) / 256, 256, 0, stream>>>(src, wb + dsto, Kact, total);
  };
  conv(ei_w1, o_eiw1, 257, 9);
  conv(ei_w2, o_eiw2, 128, 4);
  for (int l = 0; l < L; ++l){
    conv(eu_w1 + (size_t)l * 385 * DHID, o_euw1[l], 385, 13);
    conv(eu_w2 + (size_t)l * 128 * DHID, o_euw2[l], 128, 4);
    conv(mp_w1 + (size_t)l * 257 * DHID, o_mpw1[l], 257, 9);
    conv(mp_w2 + (size_t)l * 128 * DHID, o_mpw2[l], 128, 4);
    conv(nu_w1 + (size_t)l * 128 * DHID, o_nuw1[l], 128, 4);
    conv(nu_w2 + (size_t)l * 128 * DHID, o_nuw2[l], 128, 4);
  }

  float* HA = (float*)d_out;                 // output h region (also working buffer)
  float* ES = HA + (size_t)N * DHID;         // output edge_state region (in-place)
  hipMemcpyAsync(HA, h, (size_t)N * DHID * sizeof(float), hipMemcpyDeviceToDevice, stream);

  const int eblocks = (E + 15) / 16;
  const int nblocks = (N + 15) / 16;

  edge_init_kernel<<<eblocks, 256, 0, stream>>>(h, ES, snd, rcv, elen,
      wb + o_eiw1, ei_b1, wb + o_eiw2, ei_b2, E);

  float* cur = HA; float* nxt = H1;
  for (int l = 0; l < L; ++l){
    // nxt starts as a copy of cur; edge atomics accumulate messages into it
    hipMemcpyAsync(nxt, cur, (size_t)N * DHID * sizeof(float), hipMemcpyDeviceToDevice, stream);
    edge_layer_kernel<<<eblocks, 256, 0, stream>>>(cur, ES, nxt, snd, rcv, elen,
        wb + o_euw1[l], eu_b1 + (size_t)l * DHID, wb + o_euw2[l], eu_b2 + (size_t)l * DHID,
        wb + o_mpw1[l], mp_b1 + (size_t)l * DHID, wb + o_mpw2[l], mp_b2 + (size_t)l * DHID, E);
    node_kernel<<<nblocks, 256, 0, stream>>>(nxt,
        wb + o_nuw1[l], nu_b1 + (size_t)l * DHID, wb + o_nuw2[l], nu_b2 + (size_t)l * DHID, N);
    float* t = cur; cur = nxt; nxt = t;
  }
  if (cur != HA)
    hipMemcpyAsync(HA, cur, (size_t)N * DHID * sizeof(float), hipMemcpyDeviceToDevice, stream);
}
